// GRU_23081154248766
// MI455X (gfx1250) — compile-verified, matrix operand order checked by reading, not run on hardware
//
#include <hip/hip_runtime.h>

// ---------------------------------------------------------------------------
// GRU4Rec single step for MI455X (gfx1250, wave32, WMMA).
//
// V=100000 vocab, H=256 hidden, B=512 batch, L=1.
// Dominant cost: logit = tanh(h_new @ w_out^T + b_out)
//   read w_out 102.4 MB + write logit 204.8 MB  => ~13 us @ 23.3 TB/s.
// The 26.2 GFLOP matmul runs on the bf16 WMMA pipe (the only pipe fast
// enough to stay under the HBM floor); fp32 accuracy recovered with a bf16
// hi/lo split (3 WMMAs: hi*hi + lo*hi + hi*lo, f32 accumulate).
//
// Phase-2 pipeline (per block, 8 waves, double-buffered LDS, 1 barrier/iter):
//   iter i: prefetch tile i+1 into WGP$ (no registers held)
//           24 WMMAs from LDS buf[i&1]     (full reg headroom for ds pipelining)
//           issue tile i+1 staging loads   (hit WGP$/L2 thanks to prefetch)
//           epilogue tanh+store of tile i  (overlaps staging-load latency)
//           convert + store -> LDS buf[(i+1)&1]
//           __syncthreads()
// ---------------------------------------------------------------------------

typedef __attribute__((ext_vector_type(16))) __bf16         v16bf;
typedef __attribute__((ext_vector_type(8)))  float          v8f;
typedef __attribute__((ext_vector_type(4)))  float          v4f;
typedef __attribute__((ext_vector_type(8)))  unsigned short v8u16;

union Frag {                 // one 16x32 bf16 WMMA operand, per-lane view
  v16bf          bf;         // what the builtin wants
  v8u16          h[2];       // two 16B chunks (b128 loads)
  unsigned short s[16];      // element access
};

constexpr int V_SZ = 100000;
constexpr int H_SZ = 256;
constexpr int B_SZ = 512;
constexpr int NTILES_N     = V_SZ / 16;  // 6250 (exact)
constexpr int NT_PER_BLOCK = 10;         // N-tiles per block (6250 = 625*10)
constexpr int KP           = 264;        // padded LDS row stride (shorts); 132 dwords -> stride%64banks==4

__device__ __forceinline__ unsigned short f2bf_rne(float f) {
  unsigned int u = __float_as_uint(f);
  u += 0x7FFFu + ((u >> 16) & 1u);       // round-to-nearest-even to bf16
  return (unsigned short)(u >> 16);
}
__device__ __forceinline__ float bf2f(unsigned short h) {
  return __uint_as_float((unsigned int)h << 16);
}
__device__ __forceinline__ float fast_tanh(float x) {
  // tanh(x) = 1 - 2/(exp2(2x*log2e)+1); exp2/rcp are TRANS ops, overlap with mem
  float e = __builtin_amdgcn_exp2f(x * 2.8853900817779268f);
  return 1.0f - 2.0f * __builtin_amdgcn_rcpf(e + 1.0f);
}
__device__ __forceinline__ float fast_sigmoid(float x) {
  float e = __builtin_amdgcn_exp2f(-x * 1.4426950408889634f);
  return __builtin_amdgcn_rcpf(1.0f + e);
}

// ---------------------------------------------------------------------------
// Phase 1: GRU cell. One block per batch row, one thread per hidden unit.
// gi is a column gather of w_ih; gh is a 256-long dot per gate (w_hh L2-hot).
// Writes h_new into the tail of d_out (it IS the second output).
// ---------------------------------------------------------------------------
__global__ __launch_bounds__(H_SZ) void gru_cell_kernel(
    const int*   __restrict__ input,
    const float* __restrict__ hidden,
    const float* __restrict__ w_ih,
    const float* __restrict__ w_hh,
    const float* __restrict__ b_ih,
    const float* __restrict__ b_hh,
    float*       __restrict__ h_out)
{
  __shared__ float hsh[H_SZ];
  const int b = blockIdx.x;
  const int t = threadIdx.x;
  hsh[t] = hidden[b * H_SZ + t];           // layer 0 of (L,B,H)
  __syncthreads();

  const int col = input[b];
  float gi_r = w_ih[(size_t)t            * V_SZ + col] + b_ih[t];
  float gi_z = w_ih[(size_t)(H_SZ + t)   * V_SZ + col] + b_ih[H_SZ + t];
  float gi_n = w_ih[(size_t)(2*H_SZ + t) * V_SZ + col] + b_ih[2*H_SZ + t];

  float gh_r = b_hh[t], gh_z = b_hh[H_SZ + t], gh_n = b_hh[2*H_SZ + t];
  const float* wr = w_hh + (size_t)t            * H_SZ;
  const float* wz = w_hh + (size_t)(H_SZ + t)   * H_SZ;
  const float* wn = w_hh + (size_t)(2*H_SZ + t) * H_SZ;
#pragma unroll 4
  for (int k = 0; k < H_SZ; ++k) {
    float hk = hsh[k];                     // LDS broadcast, no conflicts
    gh_r = fmaf(hk, wr[k], gh_r);
    gh_z = fmaf(hk, wz[k], gh_z);
    gh_n = fmaf(hk, wn[k], gh_n);
  }
  float r = fast_sigmoid(gi_r + gh_r);
  float z = fast_sigmoid(gi_z + gh_z);
  float n = fast_tanh(gi_n + r * gh_n);
  float h = hsh[t];
  h_out[b * H_SZ + t] = (1.0f - z) * n + z * h;
}

// ---------------------------------------------------------------------------
// Phase 2 helpers: cooperative staging of one 16x256 f32 w_out tile.
// Thread t handles row nbase+t/16, k = (t%16)*16 .. +15 (64B contiguous;
// the whole tile is one contiguous 16 KB span of w_out).
// ---------------------------------------------------------------------------
__device__ __forceinline__ void stage_load(const float* __restrict__ src,
                                           float w[16]) {
  v4f f0 = *(const v4f*)(src);
  v4f f1 = *(const v4f*)(src + 4);
  v4f f2 = *(const v4f*)(src + 8);
  v4f f3 = *(const v4f*)(src + 12);
#pragma unroll
  for (int j = 0; j < 4; ++j) {
    w[j] = f0[j]; w[4 + j] = f1[j]; w[8 + j] = f2[j]; w[12 + j] = f3[j];
  }
}
__device__ __forceinline__ void stage_store(const float w[16],
                                            unsigned short* hiRow,
                                            unsigned short* loRow) {
  unsigned int* hiOut = (unsigned int*)hiRow;
  unsigned int* loOut = (unsigned int*)loRow;
#pragma unroll
  for (int p = 0; p < 8; ++p) {
    float x0 = w[2 * p], x1 = w[2 * p + 1];
    unsigned short h0 = f2bf_rne(x0), h1 = f2bf_rne(x1);
    unsigned short l0 = f2bf_rne(x0 - bf2f(h0));
    unsigned short l1 = f2bf_rne(x1 - bf2f(h1));
    hiOut[p] = (unsigned int)h0 | ((unsigned int)h1 << 16);
    loOut[p] = (unsigned int)l0 | ((unsigned int)l1 << 16);
  }
}

// ---------------------------------------------------------------------------
// Phase 2: logit = tanh(h_new @ w_out^T + b_out) via bf16 hi/lo WMMA.
//
// Grid: blockIdx & 3 -> which group of 8 M-tiles (32 total, 1 per wave),
//       blockIdx >> 2 -> chunk of NT_PER_BLOCK consecutive N-tiles.
//
// Layouts (ISA 7.12.2, wave32):
//   A 16x32 bf16 : lane l -> M = l&15; elem j<8 -> K=(l>>4)*8+j,
//                  j>=8 -> K=16+(l>>4)*8+(j-8)
//   B 32x16 bf16 : lane l -> N = l&15; same K map (B[k][n] = w_out[n][k])
//   C/D 16x16 f32: lane l -> N = l&15; VGPR r -> M = (l>>4)*8 + r
// ---------------------------------------------------------------------------
__global__ __launch_bounds__(256, 1) void logits_wmma_kernel(
    const float* __restrict__ hnew,
    const float* __restrict__ w_out,
    const float* __restrict__ b_out,
    float*       __restrict__ logit)
{
  __shared__ unsigned short BhiS[2][16 * KP];   // double-buffered bf16-hi tile
  __shared__ unsigned short BloS[2][16 * KP];   // double-buffered bf16-lo tile

  const int tid    = threadIdx.x;
  const int wave   = tid >> 5;
  const int lane   = tid & 31;
  const int l15    = lane & 15;
  const int lhalf  = lane >> 4;
  const int mt     = (blockIdx.x & 3) * 8 + wave;          // 0..31
  const int mbase  = mt * 16;
  const int ntbase = (blockIdx.x >> 2) * NT_PER_BLOCK;

  // ---- split A (h_new) into bf16 hi/lo fragments, kept in VGPRs ----------
  Frag Ahi[8], Alo[8];
  {
    const float* arow = hnew + (size_t)(mbase + l15) * H_SZ;
#pragma unroll
    for (int kk = 0; kk < 8; ++kk) {
      const int c0 = kk * 32 + lhalf * 8;      // chunks [c0,c0+8) and [c0+16,c0+24)
      v4f f0 = *(const v4f*)(arow + c0);
      v4f f1 = *(const v4f*)(arow + c0 + 4);
      v4f f2 = *(const v4f*)(arow + c0 + 16);
      v4f f3 = *(const v4f*)(arow + c0 + 20);
      float a[16];
#pragma unroll
      for (int j = 0; j < 4; ++j) {
        a[j] = f0[j]; a[4 + j] = f1[j]; a[8 + j] = f2[j]; a[12 + j] = f3[j];
      }
#pragma unroll
      for (int j = 0; j < 16; ++j) {
        unsigned short hb = f2bf_rne(a[j]);
        Ahi[kk].s[j] = hb;
        Alo[kk].s[j] = f2bf_rne(a[j] - bf2f(hb));
      }
    }
  }

  const int srow = tid >> 4;              // staging: row 0..15
  const int sks  = (tid & 15) << 4;       // staging: k 0..240 step 16
  const int soff = srow * KP + sks;       // staging LDS offset (shorts)

  // ---- prologue: stage tile 0 into buffer 0 ------------------------------
  {
    float w[16];
    stage_load(w_out + (size_t)(ntbase * 16 + srow) * H_SZ + sks, w);
    stage_store(w, &BhiS[0][soff], &BloS[0][soff]);
  }
  __syncthreads();

  for (int it = 0; it < NT_PER_BLOCK; ++it) {
    const int cur   = it & 1;
    const int nxt   = cur ^ 1;
    const int nbase = (ntbase + it) * 16;
    const bool have_next = (it + 1 < NT_PER_BLOCK);

    // ---- prefetch next tile into the near (WGP) cache: zero register cost
    if (have_next) {
      __builtin_prefetch(w_out + (size_t)(nbase + 16 + srow) * H_SZ + sks, 0, 3);
    }

    // ---- 16x16 tile: 8 k-steps x 3 WMMAs (hi*hi, lo*hi, hi*lo) ----------
    const unsigned short* bh = &BhiS[cur][0];
    const unsigned short* bl = &BloS[cur][0];
    v8f acc0 = {};                        // hi*hi chain
    v8f acc1 = {};                        // cross-term chain (independent)
#pragma unroll
    for (int kk = 0; kk < 8; ++kk) {
      const int base = l15 * KP + kk * 32 + lhalf * 8;
      Frag Bhi, Blo;
      Bhi.h[0] = *(const v8u16*)&bh[base];
      Bhi.h[1] = *(const v8u16*)&bh[base + 16];
      Blo.h[0] = *(const v8u16*)&bl[base];
      Blo.h[1] = *(const v8u16*)&bl[base + 16];
      acc0 = __builtin_amdgcn_wmma_f32_16x16x32_bf16(
                 false, Ahi[kk].bf, false, Bhi.bf, (short)0, acc0, false, false);
      acc1 = __builtin_amdgcn_wmma_f32_16x16x32_bf16(
                 false, Alo[kk].bf, false, Bhi.bf, (short)0, acc1, false, false);
      acc1 = __builtin_amdgcn_wmma_f32_16x16x32_bf16(
                 false, Ahi[kk].bf, false, Blo.bf, (short)0, acc1, false, false);
    }

    // ---- issue next tile's staging loads (should hit WGP$/L2) ------------
    float wnext[16];
    if (have_next) {
      stage_load(w_out + (size_t)(nbase + 16 + srow) * H_SZ + sks, wnext);
    }

    // ---- epilogue: + b_out, tanh, store; overlaps staging-load latency ---
    {
      const int n    = nbase + l15;
      const float bo = b_out[n];
#pragma unroll
      for (int r = 0; r < 8; ++r) {
        const int row = mbase + lhalf * 8 + r;
        logit[(size_t)row * V_SZ + n] = fast_tanh(acc0[r] + acc1[r] + bo);
      }
    }

    // ---- convert + store next tile into the other buffer -----------------
    if (have_next) {
      stage_store(wnext, &BhiS[nxt][soff], &BloS[nxt][soff]);
    }
    __syncthreads();                      // one barrier per iteration
  }
}

// ---------------------------------------------------------------------------
extern "C" void kernel_launch(void* const* d_in, const int* in_sizes, int n_in,
                              void* d_out, int out_size, void* d_ws, size_t ws_size,
                              hipStream_t stream)
{
  (void)in_sizes; (void)n_in; (void)out_size; (void)d_ws; (void)ws_size;

  const int*   input  = (const int*)  d_in[0];
  // d_in[1] = target (unused by forward)
  const float* hidden = (const float*)d_in[2];
  const float* w_ih   = (const float*)d_in[3];
  const float* w_hh   = (const float*)d_in[4];
  const float* b_ih   = (const float*)d_in[5];
  const float* b_hh   = (const float*)d_in[6];
  const float* w_out  = (const float*)d_in[7];
  const float* b_out  = (const float*)d_in[8];

  float* logit = (float*)d_out;                       // (B, V)
  float* h_new = logit + (size_t)B_SZ * V_SZ;         // (L=1, B, H) tail

  gru_cell_kernel<<<B_SZ, H_SZ, 0, stream>>>(input, hidden, w_ih, w_hh,
                                             b_ih, b_hh, h_new);

  const int nchunks = NTILES_N / NT_PER_BLOCK;        // 625
  logits_wmma_kernel<<<4 * nchunks, 256, 0, stream>>>(h_new, w_out, b_out, logit);
}
